// GeneratorRNN_19593640804694
// MI455X (gfx1250) — matrix operand order repaired and checked by reading, not compiled, not run on hardware
//
#include <hip/hip_runtime.h>
#include <hip/hip_bf16.h>
#include <math.h>

typedef __bf16 bf16_t;
typedef __bf16 v16bf __attribute__((ext_vector_type(16)));
typedef __bf16 v8bf  __attribute__((ext_vector_type(8)));
typedef float  v8f   __attribute__((ext_vector_type(8)));

#define B_  128
#define S_  1024
#define F_  512
#define H_  32
#define G_  128
#define TOK (B_ * S_)

__device__ __forceinline__ v8f zero8() {
  v8f z = {0.f, 0.f, 0.f, 0.f, 0.f, 0.f, 0.f, 0.f};
  return z;
}

__device__ __forceinline__ v8f wmma_bf16(v16bf a, v16bf b, v8f c) {
  return __builtin_amdgcn_wmma_f32_16x16x32_bf16(false, a, false, b, (short)0, c,
                                                 false, false);
}

// ---- fast activations: single TRANS-pipe ops on gfx1250 ----
__device__ __forceinline__ float fast_tanh(float x) {
#if __has_builtin(__builtin_amdgcn_tanhf)
  return __builtin_amdgcn_tanhf(x);
#elif __has_builtin(__builtin_amdgcn_tanh_f32)
  return __builtin_amdgcn_tanh_f32(x);
#else
  return tanhf(x);
#endif
}

__device__ __forceinline__ float fast_exp2(float x) {
#if __has_builtin(__builtin_amdgcn_exp2f)
  return __builtin_amdgcn_exp2f(x);
#else
  return exp2f(x);
#endif
}

__device__ __forceinline__ float fast_rcp(float x) {
#if __has_builtin(__builtin_amdgcn_rcpf)
  return __builtin_amdgcn_rcpf(x);
#else
  return 1.f / x;
#endif
}

__device__ __forceinline__ float fast_sigmoid(float x) {
  // sigmoid(x) = 1 / (1 + 2^(-x*log2(e)))
  return fast_rcp(1.f + fast_exp2(x * -1.44269504f));
}

// ---- A fragment (16x32): lane m<16 -> row m, K = k0+{0..7,16..23};
//      lane m+16 -> row m, K = k0+{8..15,24..31}. Two 16B runs per lane. ----
__device__ __forceinline__ v16bf a_frag_bf16(const bf16_t* base, int ld, int k0, int lane) {
  int m = lane & 15, hi = lane >> 4;
  const bf16_t* r = base + (size_t)m * ld + k0 + hi * 8;
  v8bf lo = *(const v8bf*)(r);
  v8bf hi8 = *(const v8bf*)(r + 16);
  return __builtin_shufflevector(lo, hi8, 0, 1, 2, 3, 4, 5, 6, 7,
                                 8, 9, 10, 11, 12, 13, 14, 15);
}

__device__ __forceinline__ v16bf a_frag_f32(const float* base, int ld, int k0, int lane) {
  int m = lane & 15, hi = lane >> 4;
  const float* r = base + (size_t)m * ld + k0 + hi * 8;
  float4 f0 = *(const float4*)(r + 0);
  float4 f1 = *(const float4*)(r + 4);
  float4 f2 = *(const float4*)(r + 16);
  float4 f3 = *(const float4*)(r + 20);
  v16bf a;
  a[0] = (bf16_t)f0.x;  a[1] = (bf16_t)f0.y;  a[2] = (bf16_t)f0.z;  a[3] = (bf16_t)f0.w;
  a[4] = (bf16_t)f1.x;  a[5] = (bf16_t)f1.y;  a[6] = (bf16_t)f1.z;  a[7] = (bf16_t)f1.w;
  a[8] = (bf16_t)f2.x;  a[9] = (bf16_t)f2.y;  a[10] = (bf16_t)f2.z; a[11] = (bf16_t)f2.w;
  a[12] = (bf16_t)f3.x; a[13] = (bf16_t)f3.y; a[14] = (bf16_t)f3.z; a[15] = (bf16_t)f3.w;
  return a;
}

// ---- B fragment direct from fp32 global row-major W[K][N] (build-once use) ----
__device__ __forceinline__ v16bf b_frag_f32(const float* W, int ldn, int k0, int n, int lane) {
  int hi = lane >> 4;
  const float* p = W + (size_t)(k0 + hi * 8) * ldn + n;
  v16bf b;
#pragma unroll
  for (int e = 0; e < 16; ++e) b[e] = (bf16_t)p[(size_t)((e & 7) + (e >> 3) * 16) * ldn];
  return b;
}

// ---- Swizzled LDS weight store: fragment-major, lane-contiguous 32B ----
template <int KTOT, int N>
__device__ __forceinline__ void stage_swizzled(bf16_t* dst, const float* W,
                                               int tid, int nthreads) {
  const int KC = KTOT / 32;
  for (int i = tid; i < KTOT * N; i += nthreads) {
    int e = i & 15;
    int lane = (i >> 4) & 31;
    int frag = i >> 9;
    int kc = frag % KC;
    int nt = frag / KC;
    int n = nt * 16 + (lane & 15);
    int k = kc * 32 + (lane >> 4) * 8 + (e & 7) + (e >> 3) * 16;
    dst[i] = (bf16_t)W[(size_t)k * N + n];
  }
}

__device__ __forceinline__ v16bf b_frag_sw(const bf16_t* sW, int kchunks, int nt, int kc,
                                           int lane) {
  const v16bf* p = (const v16bf*)sW;     // 32B per lane -> 2x ds_load_b128
  return p[(nt * kchunks + kc) * 32 + lane];
}

__device__ __forceinline__ void wave_lds_fence() {
  asm volatile("s_wait_dscnt 0\n\ts_wait_storecnt 0" ::: "memory");
}

// ---------------------------------------------------------------------------
// Kernel 1: xg = tanh(tanh(x@W1+b1)@W2+b2)@Wx + bl        [TOK,128] fp32
// ---------------------------------------------------------------------------
__global__ __launch_bounds__(128) void k1_front_mlp(
    const float* __restrict__ x, const float* __restrict__ W1, const float* __restrict__ b1,
    const float* __restrict__ W2, const float* __restrict__ b2,
    const float* __restrict__ Wx, const float* __restrict__ bl,
    float* __restrict__ xg) {
  __shared__ __align__(32) bf16_t sW1[F_ * 32];    // 32 frags, 32 KB
  __shared__ __align__(32) bf16_t sWx[64 * G_];    // 16 frags, 16 KB
  __shared__ __align__(16) bf16_t sH[4 * 16 * 64]; // per-wave scratch, 8 KB

  int tid = threadIdx.x;
  stage_swizzled<F_, 32>(sW1, W1, tid, 128);
  stage_swizzled<64, G_>(sWx, Wx, tid, 128);
  __syncthreads();

  int wave = tid >> 5, lane = tid & 31;
  int m = lane & 15, hi = lane >> 4;
  bf16_t* myH = sH + wave * (16 * 64);

  v16bf bw2[4];                               // W2 fragments register-resident
#pragma unroll
  for (int nt = 0; nt < 4; ++nt) bw2[nt] = b_frag_f32(W2, 64, 0, nt * 16 + m, lane);
  float bb1a = b1[m], bb1b = b1[16 + m];
  float bb2[4];
#pragma unroll
  for (int nt = 0; nt < 4; ++nt) bb2[nt] = b2[nt * 16 + m];
  float bblv[8];
#pragma unroll
  for (int nt = 0; nt < 8; ++nt) bblv[nt] = bl[nt * 16 + m];

  for (int blk = blockIdx.x; blk < TOK / 64; blk += gridDim.x) {
    int tok0 = blk * 64 + wave * 16;
    const float* xrow = x + (size_t)tok0 * F_;
    __builtin_prefetch(xrow + (size_t)m * F_, 0, 3);

    // layer 1: [16x512]@[512x32]
    v8f h1a = zero8(), h1b = zero8();
    for (int kc = 0; kc < 16; ++kc) {
      v16bf a = a_frag_f32(xrow, F_, kc * 32, lane);
      h1a = wmma_bf16(a, b_frag_sw(sW1, 16, 0, kc, lane), h1a);
      h1b = wmma_bf16(a, b_frag_sw(sW1, 16, 1, kc, lane), h1b);
    }
#pragma unroll
    for (int r = 0; r < 8; ++r) {
      int row = r + hi * 8;
      myH[row * 32 + m]      = (bf16_t)fast_tanh(h1a[r] + bb1a);
      myH[row * 32 + 16 + m] = (bf16_t)fast_tanh(h1b[r] + bb1b);
    }
    wave_lds_fence();

    // layer 2: [16x32]@[32x64]
    v16bf a1 = a_frag_bf16(myH, 32, 0, lane);
    v8f h2[4];
#pragma unroll
    for (int nt = 0; nt < 4; ++nt) h2[nt] = wmma_bf16(a1, bw2[nt], zero8());
#pragma unroll
    for (int nt = 0; nt < 4; ++nt) {
      int n = nt * 16 + m;
#pragma unroll
      for (int r = 0; r < 8; ++r) {
        int row = r + hi * 8;
        myH[row * 64 + n] = (bf16_t)fast_tanh(h2[nt][r] + bb2[nt]);
      }
    }
    wave_lds_fence();

    // gate layer: [16x64]@[64x128]
    v16bf a2a = a_frag_bf16(myH, 64, 0, lane);
    v16bf a2b = a_frag_bf16(myH, 64, 32, lane);
    float* xgo = xg + (size_t)tok0 * G_;
#pragma unroll
    for (int nt = 0; nt < 8; ++nt) {
      int n = nt * 16 + m;
      v8f acc = wmma_bf16(a2a, b_frag_sw(sWx, 2, nt, 0, lane), zero8());
      acc     = wmma_bf16(a2b, b_frag_sw(sWx, 2, nt, 1, lane), acc);
#pragma unroll
      for (int r = 0; r < 8; ++r) {
        int row = r + hi * 8;
        xgo[(size_t)row * G_ + n] = acc[r] + bblv[nt];
      }
    }
  }
}

// ---------------------------------------------------------------------------
// Kernel 2: LSTM scan. 8 WGs x 16 batch rows; wave w owns gate cols [16w,16w+16).
// ---------------------------------------------------------------------------
__global__ __launch_bounds__(256) void k2_lstm_scan(
    const float* __restrict__ xg, const float* __restrict__ Wh,
    bf16_t* __restrict__ hs) {
  __shared__ float gates[16 * G_];
  __shared__ __align__(16) bf16_t hbuf[16 * H_];

  int tid = threadIdx.x, wave = tid >> 5, lane = tid & 31;
  int wg = blockIdx.x;
  int m = lane & 15, hi = lane >> 4;
  int n = wave * 16 + m;
  int gate = wave >> 1;   // 0=i 1=f 2=g 3=o

  v16bf bwh;              // register-resident Wh B fragment, used 1024 times
#pragma unroll
  for (int e = 0; e < 16; ++e) bwh[e] = (bf16_t)Wh[(size_t)(hi * 8 + (e & 7) + (e >> 3) * 16) * G_ + n];

  for (int i = tid; i < 16 * H_; i += 256) hbuf[i] = (bf16_t)0.f;
  float c0 = 0.f, c1 = 0.f;
  int r0 = tid >> 5, j0 = tid & 31;
  int r1 = r0 + 8;
  __syncthreads();

  const float* xg_wg = xg + (size_t)(wg * 16) * S_ * G_;
  bf16_t* hs_wg = hs + (size_t)(wg * 16) * S_ * H_;

  for (int t = 0; t < S_; ++t) {
    v16bf a = a_frag_bf16(hbuf, H_, 0, lane);
    v8f acc = wmma_bf16(a, bwh, zero8());

#pragma unroll
    for (int r = 0; r < 8; ++r) {
      int row = r + hi * 8;
      float g = acc[r] + xg_wg[((size_t)row * S_ + t) * G_ + n];
      gates[row * G_ + n] = (gate == 2) ? fast_tanh(g) : fast_sigmoid(g);
    }
    __syncthreads();

    {
      float iv = gates[r0 * G_ + j0], fv = gates[r0 * G_ + 32 + j0];
      float gv = gates[r0 * G_ + 64 + j0], ov = gates[r0 * G_ + 96 + j0];
      c0 = fv * c0 + iv * gv;
      float hv = ov * fast_tanh(c0);
      hbuf[r0 * H_ + j0] = (bf16_t)hv;
      hs_wg[((size_t)r0 * S_ + t) * H_ + j0] = (bf16_t)hv;
    }
    {
      float iv = gates[r1 * G_ + j0], fv = gates[r1 * G_ + 32 + j0];
      float gv = gates[r1 * G_ + 64 + j0], ov = gates[r1 * G_ + 96 + j0];
      c1 = fv * c1 + iv * gv;
      float hv = ov * fast_tanh(c1);
      hbuf[r1 * H_ + j0] = (bf16_t)hv;
      hs_wg[((size_t)r1 * S_ + t) * H_ + j0] = (bf16_t)hv;
    }
    __syncthreads();
  }
}

// ---------------------------------------------------------------------------
// Kernel 3: out = tanh(hs@W3+b3)@W4 + b4                  [TOK,512] fp32
// ---------------------------------------------------------------------------
__global__ __launch_bounds__(128) void k3_back_mlp(
    const bf16_t* __restrict__ hs, const float* __restrict__ W3,
    const float* __restrict__ b3, const float* __restrict__ W4,
    const float* __restrict__ b4, float* __restrict__ out) {
  extern __shared__ __align__(32) char smem3[];
  bf16_t* sW4 = (bf16_t*)smem3;        // 64 frags swizzled, 64 KB
  bf16_t* sH  = sW4 + 64 * 512;        // 4 waves * 16*64, 8 KB

  int tid = threadIdx.x;
  stage_swizzled<64, 512>(sW4, W4, tid, 128);
  __syncthreads();

  int wave = tid >> 5, lane = tid & 31;
  int m = lane & 15, hi = lane >> 4;
  bf16_t* myH = sH + wave * (16 * 64);

  v16bf bw3[4];                        // W3 fragments register-resident
#pragma unroll
  for (int nt = 0; nt < 4; ++nt) bw3[nt] = b_frag_f32(W3, 64, 0, nt * 16 + m, lane);
  float bb3[4];
#pragma unroll
  for (int nt = 0; nt < 4; ++nt) bb3[nt] = b3[nt * 16 + m];

  for (int blk = blockIdx.x; blk < TOK / 64; blk += gridDim.x) {
    int tok0 = blk * 64 + wave * 16;

    // layer 3: [16x32]@[32x64]
    v16bf a = a_frag_bf16(hs + (size_t)tok0 * H_, H_, 0, lane);
    v8f h3[4];
#pragma unroll
    for (int nt = 0; nt < 4; ++nt) h3[nt] = wmma_bf16(a, bw3[nt], zero8());
#pragma unroll
    for (int nt = 0; nt < 4; ++nt) {
      int n = nt * 16 + m;
#pragma unroll
      for (int r = 0; r < 8; ++r) {
        int row = r + hi * 8;
        myH[row * 64 + n] = (bf16_t)fast_tanh(h3[nt][r] + bb3[nt]);
      }
    }
    wave_lds_fence();

    // layer 4: [16x64]@[64x512]
    v16bf a0 = a_frag_bf16(myH, 64, 0, lane);
    v16bf a1 = a_frag_bf16(myH, 64, 32, lane);
    float* orow = out + (size_t)tok0 * 512;
    for (int grp = 0; grp < 4; ++grp) {
      v8f o[8];
#pragma unroll
      for (int nt = 0; nt < 8; ++nt) {
        int nn = grp * 8 + nt;
        v8f t0 = wmma_bf16(a0, b_frag_sw(sW4, 2, nn, 0, lane), zero8());
        o[nt]  = wmma_bf16(a1, b_frag_sw(sW4, 2, nn, 1, lane), t0);
      }
#pragma unroll
      for (int nt = 0; nt < 8; ++nt) {
        int n = (grp * 8 + nt) * 16 + m;
        float bb = b4[n];
#pragma unroll
        for (int r = 0; r < 8; ++r) {
          int row = r + hi * 8;
          orow[(size_t)row * 512 + n] = o[nt][r] + bb;
        }
      }
    }
  }
}

// ---------------------------------------------------------------------------
extern "C" void kernel_launch(void* const* d_in, const int* in_sizes, int n_in,
                              void* d_out, int out_size, void* d_ws, size_t ws_size,
                              hipStream_t stream) {
  const float* x  = (const float*)d_in[0];
  const float* W1 = (const float*)d_in[1];
  const float* b1 = (const float*)d_in[2];
  const float* W2 = (const float*)d_in[3];
  const float* b2 = (const float*)d_in[4];
  const float* Wx = (const float*)d_in[5];
  const float* Wh = (const float*)d_in[6];
  const float* bl = (const float*)d_in[7];
  const float* W3 = (const float*)d_in[8];
  const float* b3 = (const float*)d_in[9];
  const float* W4 = (const float*)d_in[10];
  const float* b4 = (const float*)d_in[11];
  float* out = (float*)d_out;

  float*  xg = (float*)d_ws;                                              // 64 MB
  bf16_t* hs = (bf16_t*)((char*)d_ws + (size_t)TOK * G_ * sizeof(float)); // 8 MB

  k1_front_mlp<<<512, 128, 0, stream>>>(x, W1, b1, W2, b2, Wx, bl, xg);
  k2_lstm_scan<<<8, 256, 0, stream>>>(xg, Wh, hs);
  k3_back_mlp<<<512, 128, (64 * 512 + 4 * 16 * 64) * (int)sizeof(bf16_t), stream>>>(
      hs, W3, b3, W4, b4, out);
}
